// AmacrineCellFeedback_7481833030195
// MI455X (gfx1250) — matrix operand order repaired and checked by reading, not compiled, not run on hardware
//
#include <hip/hip_runtime.h>
#include <hip/hip_bf16.h>
#include <math.h>

// ---------------------------------------------------------------------------
// Problem constants (from reference)
// ---------------------------------------------------------------------------
#define N_AC 8192
#define N_BC 8192
#define T_K  52

typedef __attribute__((ext_vector_type(16))) __bf16 v16bf;
typedef __attribute__((ext_vector_type(8)))  float  v8f;
typedef __attribute__((ext_vector_type(4)))  float  v4f;

// Fragment buffer: for each k-step (K chunk of 32) and each of the 4 N-tiles,
// the 32-lane x 16-half B fragment is stored contiguously (lane-major, 32 B
// per lane -> two global_load_b128 per fragment load, L2-resident, 1 MB).
#define K_STEPS    (N_BC / 32)                         // 256
#define FRAG_HALFS ((size_t)K_STEPS * 4 * 32 * 16)     // 524288 bf16 = 1 MB

// K-split across workgroups (grid.y) and waves: more waves in flight so the
// aggregate outstanding-load pool can cover the 23.3 TB/s x ~600ns = ~14 MB
// bandwidth-delay product of HBM.
#define KSPLIT_WG   4
#define WAVES_PER_WG 4
#define K_PER_WAVE  (N_BC / (KSPLIT_WG * WAVES_PER_WG))   // 512
#define STEPS_PER_WAVE (K_PER_WAVE / 32)                  // 16

// ---------------------------------------------------------------------------
// Kernel 0: swizzle x (N_BC x T_K f32 row-major) into bf16 WMMA B-fragments.
// B layout for V_WMMA_F32_16X16X32_BF16 (32x16, KxN):
//   lanes 0-15 : column N = lane,    half idx -> K {0..7} U {16..23}
//   lanes 16-31: column N = lane-16, half idx -> K {8..15} U {24..31}
// Linear half index: ((kstep*4 + tile)*32 + lane)*16 + idx.  t>=52 -> 0 pad.
// ---------------------------------------------------------------------------
__global__ void build_xfrag_kernel(const float* __restrict__ x,
                                   __bf16* __restrict__ frag) {
    int tid   = blockIdx.x * blockDim.x + threadIdx.x;   // one bf16 per thread
    int idx   = tid & 15;
    int lane  = (tid >> 4) & 31;
    int tile  = (tid >> 9) & 3;
    int kstep = tid >> 11;
    int K = kstep * 32 + (lane >> 4) * 8 + (idx & 7) + (idx >> 3) * 16;
    int t = tile * 16 + (lane & 15);
    float v = (t < T_K) ? x[(size_t)K * T_K + t] : 0.0f;
    frag[tid] = (__bf16)v;
}

// ---------------------------------------------------------------------------
// Kernel 1: zero the per-row f32 partial-sum accumulator (graph-safe init).
// ---------------------------------------------------------------------------
__global__ void zero_sacc_kernel(float* __restrict__ s_acc) {
    int i = blockIdx.x * blockDim.x + threadIdx.x;
    if (i < N_AC) s_acc[i] = 0.0f;
}

// ---------------------------------------------------------------------------
// Kernel 2: fused GEMM + kernel-weighted temporal reduction.
// Grid: (512 row-blocks, 4 K-chunks) x 128 threads (4 waves).
// Wave owns rows [16*bx, 16*bx+16) x K slice of 512, in 16 k-steps of 32,
// manually unrolled 2x with independent register sets so the 8 WMMAs of a
// pair issue behind one batched load clause.
// ---------------------------------------------------------------------------
__global__ void __launch_bounds__(128)
gemm_ac_kernel(const float* __restrict__ W,
               const __bf16* __restrict__ frag,
               const float* __restrict__ ack,
               float* __restrict__ s_acc) {
    const int lane = threadIdx.x & 31;
    const int wave = threadIdx.x >> 5;
    const int rowBase = blockIdx.x * 16;

    // A-operand addressing (16-bit A 16x32 layout):
    //   lanes 0-15 : M = lane,    K offsets {0..7} U {16..23}  (seg = 0)
    //   lanes 16-31: M = lane-16, K offsets {8..15} U {24..31} (seg = 8)
    const int row = rowBase + (lane & 15);
    const int seg = (lane >> 4) * 8;
    const int kBase = blockIdx.y * (N_BC / KSPLIT_WG) + wave * K_PER_WAVE;

    const float* __restrict__ wrow = W + (size_t)row * N_BC + kBase + seg;
    const v16bf* __restrict__ fragv = (const v16bf*)frag;
    const size_t stepBase = (size_t)(kBase >> 5) * 4;   // fragment index base

    v8f c0 = {}, c1 = {}, c2 = {}, c3 = {};

    for (int step = 0; step < STEPS_PER_WAVE; step += 2) {
        const float* wp0 = wrow + step * 32;
        const float* wp1 = wp0 + 32;
        // Prefetch the W stream ~16 k-steps (2 KB) ahead into near cache.
        __builtin_prefetch(wp0 + 512, 0, 3);

        // ---- batched loads for the step pair ----
        v4f e0 = *(const v4f*)(wp0);        // A(step)   K = seg+0..3
        v4f e1 = *(const v4f*)(wp0 + 4);    //           K = seg+4..7
        v4f e2 = *(const v4f*)(wp0 + 16);   //           K = seg+16..19
        v4f e3 = *(const v4f*)(wp0 + 20);   //           K = seg+20..23
        v4f g0 = *(const v4f*)(wp1);        // A(step+1)
        v4f g1 = *(const v4f*)(wp1 + 4);
        v4f g2 = *(const v4f*)(wp1 + 16);
        v4f g3 = *(const v4f*)(wp1 + 20);

        const size_t fb0 = (stepBase + (size_t)step * 4) * 32 + lane;
        const size_t fb1 = fb0 + 4 * 32;
        v16bf b00 = fragv[fb0];
        v16bf b01 = fragv[fb0 + 32];
        v16bf b02 = fragv[fb0 + 64];
        v16bf b03 = fragv[fb0 + 96];
        v16bf b10 = fragv[fb1];
        v16bf b11 = fragv[fb1 + 32];
        v16bf b12 = fragv[fb1 + 64];
        v16bf b13 = fragv[fb1 + 96];

        // ---- in-register f32 -> bf16 conversion (v_cvt_pk_bf16_f32) ----
        v16bf a0, a1;
        #pragma unroll
        for (int j = 0; j < 4; ++j) {
            a0[j]      = (__bf16)e0[j];
            a0[j + 4]  = (__bf16)e1[j];
            a0[j + 8]  = (__bf16)e2[j];
            a0[j + 12] = (__bf16)e3[j];
            a1[j]      = (__bf16)g0[j];
            a1[j + 4]  = (__bf16)g1[j];
            a1[j + 8]  = (__bf16)g2[j];
            a1[j + 12] = (__bf16)g3[j];
        }

        // ---- 8 WMMAs for the pair ----
        c0 = __builtin_amdgcn_wmma_f32_16x16x32_bf16(false, a0, false, b00,
                                                     (short)0, c0, false, false);
        c1 = __builtin_amdgcn_wmma_f32_16x16x32_bf16(false, a0, false, b01,
                                                     (short)0, c1, false, false);
        c2 = __builtin_amdgcn_wmma_f32_16x16x32_bf16(false, a0, false, b02,
                                                     (short)0, c2, false, false);
        c3 = __builtin_amdgcn_wmma_f32_16x16x32_bf16(false, a0, false, b03,
                                                     (short)0, c3, false, false);
        c0 = __builtin_amdgcn_wmma_f32_16x16x32_bf16(false, a1, false, b10,
                                                     (short)0, c0, false, false);
        c1 = __builtin_amdgcn_wmma_f32_16x16x32_bf16(false, a1, false, b11,
                                                     (short)0, c1, false, false);
        c2 = __builtin_amdgcn_wmma_f32_16x16x32_bf16(false, a1, false, b12,
                                                     (short)0, c2, false, false);
        c3 = __builtin_amdgcn_wmma_f32_16x16x32_bf16(false, a1, false, b13,
                                                     (short)0, c3, false, false);
    }

    // Epilogue: C/D 16x16 f32 layout: VGPR v, lanes 0-15 -> (M=v, N=lane),
    // lanes 16-31 -> (M=v+8, N=lane-16).  Fold ac_kernel, reduce over N.
    const int n  = lane & 15;
    const int mb = (lane >> 4) * 8;
    #pragma unroll
    for (int v = 0; v < 8; ++v) {
        const int m = mb + v;
        const float* kr = ack + (size_t)(rowBase + m) * T_K;
        float acc = kr[n] * c0[v];                          // t =  0..15
        acc += kr[16 + n] * c1[v];                          // t = 16..31
        acc += kr[32 + n] * c2[v];                          // t = 32..47
        if (48 + n < T_K) acc += kr[48 + n] * c3[v];        // t = 48..51
        // Reduce across the 16 lanes of this N-group (masks <16 keep groups).
        acc += __shfl_xor(acc, 1, 32);
        acc += __shfl_xor(acc, 2, 32);
        acc += __shfl_xor(acc, 4, 32);
        acc += __shfl_xor(acc, 8, 32);
        if (n == 0) atomicAdd(&s_acc[rowBase + m], acc);
    }
}

// ---------------------------------------------------------------------------
// Kernel 3: per-cell sigmoid  out = 1 / (1 + exp(-slope * (s - offset)))
// ---------------------------------------------------------------------------
__global__ void sigmoid_kernel(const float* __restrict__ s_acc,
                               const float* __restrict__ slope,
                               const float* __restrict__ offset,
                               float* __restrict__ out) {
    int i = blockIdx.x * blockDim.x + threadIdx.x;
    if (i < N_AC) {
        float z = slope[i] * (s_acc[i] - offset[i]);
        out[i] = 1.0f / (1.0f + __expf(-z));
    }
}

// ---------------------------------------------------------------------------
// kernel_launch: inputs in setup_inputs() order:
//   d_in[0] = x (N_BC x T_K f32), d_in[1] = bc_ac_weight (N_AC x N_BC f32),
//   d_in[2] = ac_kernel (N_AC x T_K f32), d_in[3] = slope, d_in[4] = offset.
// Workspace layout: [0, 32KB) f32 s_acc[8192]; [32KB, 32KB+1MB) bf16 x-frags.
// ---------------------------------------------------------------------------
extern "C" void kernel_launch(void* const* d_in, const int* in_sizes, int n_in,
                              void* d_out, int out_size, void* d_ws, size_t ws_size,
                              hipStream_t stream) {
    const float* x      = (const float*)d_in[0];
    const float* W      = (const float*)d_in[1];
    const float* ack    = (const float*)d_in[2];
    const float* slope  = (const float*)d_in[3];
    const float* offset = (const float*)d_in[4];
    float* out = (float*)d_out;

    float*  s_acc = (float*)d_ws;
    __bf16* frag  = (__bf16*)((char*)d_ws + 32768);

    (void)in_sizes; (void)n_in; (void)out_size; (void)ws_size;

    // 1) swizzle x into bf16 WMMA B-fragments (L2-resident, 1 MB)
    build_xfrag_kernel<<<(int)(FRAG_HALFS / 256), 256, 0, stream>>>(x, frag);
    // 2) zero the K-split accumulator
    zero_sacc_kernel<<<N_AC / 256, 256, 0, stream>>>(s_acc);
    // 3) fused GEMM + temporal reduction: 512 row-blocks x 4 K-chunks
    dim3 grid(N_AC / 16, KSPLIT_WG);
    gemm_ac_kernel<<<grid, WAVES_PER_WG * 32, 0, stream>>>(W, frag, ack, s_acc);
    // 4) per-cell sigmoid
    sigmoid_kernel<<<N_AC / 256, 256, 0, stream>>>(s_acc, slope, offset, out);
}
